// S2ESNet_75505525064523
// MI455X (gfx1250) — matrix operand order repaired and checked by reading, not compiled, not run on hardware
//
#include <hip/hip_runtime.h>

typedef __attribute__((ext_vector_type(16))) _Float16 v16h;
typedef __attribute__((ext_vector_type(8)))  float    v8f;

#define HW 169
#define BSZ 1024

// ---------------------------------------------------------------------------
// BN-fold + pack conv weights into WMMA A-fragment order (16-bit A 16x32).
// A layout: lane l: m = l&15, khalf = l>>4; element j: K = group*16 + khalf*8 + (j&7), group=j>>3
// Fragment storage: apack[((tileM*nchunks + kc)*32 + lane)*16 + j]
// ---------------------------------------------------------------------------
__global__ void pack_weights(const float* __restrict__ w,
                             const float* __restrict__ g,
                             const float* __restrict__ v,
                             _Float16* __restrict__ apack,
                             int Cout, int K, int nchunks, int total)
{
    int idx = blockIdx.x * blockDim.x + threadIdx.x;
    if (idx >= total) return;
    int j     = idx & 15;
    int lane  = (idx >> 4) & 31;
    int t     = idx >> 9;            // tileM*nchunks + kc
    int kc    = t % nchunks;
    int tileM = t / nchunks;
    int mrow  = tileM * 16 + (lane & 15);
    int khalf = lane >> 4;
    int group = j >> 3;
    int k     = kc * 32 + group * 16 + khalf * 8 + (j & 7);
    float val = 0.f;
    if (mrow < Cout && k < K)
        val = w[(size_t)mrow * K + k] * g[mrow] * rsqrtf(v[mrow] + 1e-5f);
    apack[idx] = (_Float16)val;
}

__global__ void fold_bias(const float* b, const float* g, const float* beta,
                          const float* m, const float* v, float* bias_out, int Cout)
{
    int o = blockIdx.x * blockDim.x + threadIdx.x;
    if (o < Cout)
        bias_out[o] = (b[o] - m[o]) * g[o] * rsqrtf(v[o] + 1e-5f) + beta[o];
}

__global__ void fold_w1x1(const float* w, const float* g, const float* v,
                          float* wout, int Cout, int Cin)
{
    int idx = blockIdx.x * blockDim.x + threadIdx.x;
    if (idx < Cout * Cin) {
        int o = idx / Cin;
        wout[idx] = w[idx] * g[o] * rsqrtf(v[o] + 1e-5f);
    }
}

// ---------------------------------------------------------------------------
// Implicit-GEMM 3x3 SAME conv + bias + ReLU via v_wmma_f32_16x16x32_f16.
// One workgroup per (image, 16-pixel tile): blockDim = 32*tilesM.
// Phase 1: all threads cooperatively im2col-gather the B fragments for ALL
//          K-chunks into LDS in WMMA fragment order (done ONCE per workgroup).
// Phase 2: wave w computes out-channel tile w: contiguous 32B A-fragment load
//          (pre-packed weights) + 32B LDS B-fragment load + WMMA. No
//          divergence -> EXEC all-1s for every WMMA.
// B fragment layout (32x16 f16): lane&15 = pixel col, lane>>4 = K half,
//                                elem j -> K = (j>>3)*16 + (lane>>4)*8 + (j&7).
// C/D layout: lane&15 = col, VGPR r -> row r + 8*(lane>>4).
// ---------------------------------------------------------------------------
__global__ void conv3x3_wmma(const _Float16* __restrict__ apack,
                             const float*    __restrict__ bias,
                             const _Float16* __restrict__ in_h,  // f16 acts (or null)
                             const float*    __restrict__ in_f,  // f32 input (or null)
                             _Float16*       __restrict__ out,
                             int Cin, int Cout, int nchunks)
{
    extern __shared__ char smem_raw[];
    _Float16* bpack = (_Float16*)smem_raw;        // [nchunks*512]

    const int tileN = blockIdx.x;                 // 0..10 (11*16 >= 169)
    const int b     = blockIdx.z;
    const int tid   = threadIdx.x;
    const int K     = Cin * 9;
    const size_t in_img = (size_t)b * Cin * HW;

    // ---- Phase 1: cooperative im2col into LDS (fragment order) ----
    const int total = nchunks * 512;
    for (int idx = tid; idx < total; idx += blockDim.x) {
        int j    = idx & 15;
        int lane = (idx >> 4) & 31;
        int kc   = idx >> 9;
        int k    = kc * 32 + (j >> 3) * 16 + (lane >> 4) * 8 + (j & 7);
        int ncol = tileN * 16 + (lane & 15);
        float val = 0.f;
        if (ncol < HW && k < K) {
            int py = ncol / 13, px = ncol - py * 13;
            int ci = k / 9;
            int rs = k - ci * 9;
            int r  = rs / 3;
            int s  = rs - r * 3;
            int y = py + r - 1, x = px + s - 1;
            if ((unsigned)y < 13u && (unsigned)x < 13u) {
                size_t off = in_img + (size_t)ci * HW + y * 13 + x;
                val = in_f ? in_f[off] : (float)in_h[off];
            }
        }
        bpack[idx] = (_Float16)val;
    }
    __syncthreads();

    // ---- Phase 2: each wave owns one 16-out-channel tile ----
    const int tileM = tid >> 5;
    const int lane  = tid & 31;
    const _Float16* aptr = apack + ((size_t)tileM * nchunks * 32 + lane) * 16;
    const _Float16* bptr = bpack + lane * 16;

    v8f acc = {};
#pragma unroll 2
    for (int kc = 0; kc < nchunks; ++kc) {
        v16h afrag = *(const v16h*)(aptr + (size_t)kc * 512);
        v16h bfrag = *(const v16h*)(bptr + kc * 512);
        acc = __builtin_amdgcn_wmma_f32_16x16x32_f16(false, afrag, false, bfrag,
                                                     (short)0, acc, false, false);
    }

    const int ncol = tileN * 16 + (lane & 15);
    if (ncol >= HW) return;
    const int mbase = (lane >> 4) * 8;
#pragma unroll
    for (int r = 0; r < 8; ++r) {
        int ch = tileM * 16 + mbase + r;
        if (ch < Cout) {
            float vv = acc[r] + bias[ch];
            vv = vv > 0.f ? vv : 0.f;
            out[(size_t)b * Cout * HW + (size_t)ch * HW + ncol] = (_Float16)vv;
        }
    }
}

// ---------------------------------------------------------------------------
// 1x1 conv + folded BN + sigmoid (tiny: SAEM/SEEM T blocks)
// ---------------------------------------------------------------------------
__global__ void conv1x1_sig(const float* __restrict__ w, const float* __restrict__ bias,
                            const _Float16* __restrict__ in, _Float16* __restrict__ out,
                            int Cin, int Cout, int total)
{
    int idx = blockIdx.x * blockDim.x + threadIdx.x;
    if (idx >= total) return;
    int n = idx % HW;
    int t = idx / HW;
    int o = t % Cout;
    int b = t / Cout;
    float s = bias[o];
    const _Float16* ip = in + (size_t)b * Cin * HW + n;
    for (int ci = 0; ci < Cin; ++ci)
        s += (float)ip[(size_t)ci * HW] * w[o * Cin + ci];
    out[idx] = (_Float16)(1.f / (1.f + expf(-s)));
}

// ---------------------------------------------------------------------------
// SAEM collapsed: u[c] = sum_n drw[n]*t2[c,n];  gate[m] = sum_c t1[c,m]*u[c]
// ss1 = a * gate  -> fused channels [0..31]
// ---------------------------------------------------------------------------
__global__ __launch_bounds__(256)
void saem_gate_fuse(const _Float16* __restrict__ t1, const _Float16* __restrict__ t2,
                    const float* __restrict__ drw,
                    const _Float16* __restrict__ a3, _Float16* __restrict__ fused)
{
    __shared__ float u[16];
    __shared__ float gate[HW];
    int b = blockIdx.x, tid = threadIdx.x;
    if (tid < 16) {
        float s = 0;
        const _Float16* p = t2 + (size_t)b * 16 * HW + (size_t)tid * HW;
        for (int n = 0; n < HW; ++n) s += drw[n] * (float)p[n];
        u[tid] = s;
    }
    __syncthreads();
    if (tid < HW) {
        float g = 0;
        const _Float16* p = t1 + (size_t)b * 16 * HW + tid;
        for (int c = 0; c < 16; ++c) g += (float)p[(size_t)c * HW] * u[c];
        gate[tid] = g;
    }
    __syncthreads();
    for (int idx = tid; idx < 32 * HW; idx += 256) {
        int n = idx % HW;
        fused[(size_t)b * 64 * HW + idx] =
            (_Float16)((float)a3[(size_t)b * 32 * HW + idx] * gate[n]);
    }
}

// ---------------------------------------------------------------------------
// SEEM collapsed: u2[n] = sum_d drw[d]*t2[d,n];  gate[c] = sum_n t1[c,n]*u2[n]
// ss2 = b * gate  -> fused channels [32..63]
// ---------------------------------------------------------------------------
__global__ __launch_bounds__(256)
void seem_gate_fuse(const _Float16* __restrict__ t1, const _Float16* __restrict__ t2,
                    const float* __restrict__ drw,
                    const _Float16* __restrict__ b3, _Float16* __restrict__ fused)
{
    __shared__ float u2[HW];
    __shared__ float gate[32];
    int b = blockIdx.x, tid = threadIdx.x;
    if (tid < HW) {
        float s = 0;
        const _Float16* p = t2 + (size_t)b * 32 * HW + tid;
        for (int d = 0; d < 32; ++d) s += drw[d] * (float)p[(size_t)d * HW];
        u2[tid] = s;
    }
    __syncthreads();
    if (tid < 32) {
        float g = 0;
        const _Float16* p = t1 + (size_t)b * 32 * HW + (size_t)tid * HW;
        for (int n = 0; n < HW; ++n) g += (float)p[n] * u2[n];
        gate[tid] = g;
    }
    __syncthreads();
    for (int idx = tid; idx < 32 * HW; idx += 256) {
        int ch = idx / HW;
        fused[(size_t)b * 64 * HW + 32 * HW + idx] =
            (_Float16)((float)b3[(size_t)b * 32 * HW + idx] * gate[ch]);
    }
}

// ---------------------------------------------------------------------------
// Slot attention + pool + FC, one 256-thread workgroup per image, all in LDS.
// feats = fusionOut reinterpreted (reshape!) as [169][32].
// ---------------------------------------------------------------------------
#define SA_N 169
#define SA_D 32
#define SA_S 8

__global__ __launch_bounds__(256)
void slot_attn_kernel(const _Float16* __restrict__ feats,     // [B][32*169] raw
                      const float* __restrict__ noise,        // [B][8][32]
                      const float* ln_in_g, const float* ln_in_b,
                      const float* ln_s_g,  const float* ln_s_b,
                      const float* ln_m_g,  const float* ln_m_b,
                      const float* mu_p, const float* lsig,
                      const float* wq, const float* wk, const float* wv,
                      const float* wi, const float* wh, const float* bi, const float* bh,
                      const float* m1w, const float* m1b, const float* m2w, const float* m2b,
                      const float* fcw, const float* fcb,
                      float* __restrict__ out)
{
    __shared__ float    xbuf[SA_N * SA_D];       // xn, later reused as attn[169][8]
    __shared__ _Float16 kbuf[SA_N * SA_D];
    __shared__ _Float16 vbuf[SA_N * SA_D];
    __shared__ float slots[SA_S * SA_D], prev[SA_S * SA_D], q[SA_S * SA_D], lnb[SA_S * SA_D];
    __shared__ float gi[SA_S * 96], gh[SA_S * 96], h1[SA_S * 64];
    __shared__ float colsum[SA_S], muS[SA_S], invS[SA_S];

    const int b = blockIdx.x, tid = threadIdx.x;

    // load feats (pure reshape: flat index n*32+c)
    for (int idx = tid; idx < SA_N * SA_D; idx += 256)
        xbuf[idx] = (float)feats[(size_t)b * SA_N * SA_D + idx];
    __syncthreads();

    // LayerNorm over last dim, in place
    if (tid < SA_N) {
        float s = 0;
        for (int c = 0; c < 32; ++c) s += xbuf[tid * 32 + c];
        float mn = s * (1.f / 32.f), vr = 0;
        for (int c = 0; c < 32; ++c) { float d = xbuf[tid * 32 + c] - mn; vr += d * d; }
        float inv = rsqrtf(vr * (1.f / 32.f) + 1e-5f);
        for (int c = 0; c < 32; ++c)
            xbuf[tid * 32 + c] = (xbuf[tid * 32 + c] - mn) * inv * ln_in_g[c] + ln_in_b[c];
    }
    __syncthreads();

    // k = xn @ wk^T, v = xn @ wv^T
    for (int idx = tid; idx < SA_N * SA_D; idx += 256) {
        int n = idx >> 5, d = idx & 31;
        float sk = 0, sv = 0;
        for (int e = 0; e < 32; ++e) {
            float xv = xbuf[n * 32 + e];
            sk += xv * wk[d * 32 + e];
            sv += xv * wv[d * 32 + e];
        }
        kbuf[idx] = (_Float16)sk;
        vbuf[idx] = (_Float16)sv;
    }
    __syncthreads();

    if (tid < SA_S * SA_D) {
        int d = tid & 31;
        slots[tid] = mu_p[d] + expf(lsig[d]) * noise[(size_t)b * SA_S * SA_D + tid];
    }
    __syncthreads();

    float* attn = xbuf;                 // reuse (needs 169*8 <= 169*32)
    const float scale = 0.17677669529663689f;   // 32^-0.5

    for (int it = 0; it < 3; ++it) {
        if (tid < SA_S * SA_D) prev[tid] = slots[tid];
        __syncthreads();

        // q = ln_s(slots) @ wq^T * scale
        if (tid < SA_S) {
            float s = 0;
            for (int e = 0; e < 32; ++e) s += slots[tid * 32 + e];
            float mn = s * (1.f / 32.f), vr = 0;
            for (int e = 0; e < 32; ++e) { float d = slots[tid * 32 + e] - mn; vr += d * d; }
            muS[tid] = mn; invS[tid] = rsqrtf(vr * (1.f / 32.f) + 1e-5f);
        }
        __syncthreads();
        if (tid < SA_S * SA_D) {
            int s = tid >> 5, d = tid & 31;
            lnb[tid] = (slots[tid] - muS[s]) * invS[s] * ln_s_g[d] + ln_s_b[d];
        }
        __syncthreads();
        if (tid < SA_S * SA_D) {
            int s = tid >> 5, d = tid & 31;
            float acc = 0;
            for (int e = 0; e < 32; ++e) acc += lnb[s * 32 + e] * wq[d * 32 + e];
            q[tid] = acc * scale;
        }
        __syncthreads();

        // logits, softmax over slots, + eps
        if (tid < SA_N) {
            float l[SA_S];
            float mx = -1e30f;
            for (int s = 0; s < SA_S; ++s) {
                float acc = 0;
                for (int e = 0; e < 32; ++e) acc += (float)kbuf[tid * 32 + e] * q[s * 32 + e];
                l[s] = acc; mx = fmaxf(mx, acc);
            }
            float sum = 0;
            for (int s = 0; s < SA_S; ++s) { l[s] = expf(l[s] - mx); sum += l[s]; }
            float rs = 1.f / sum;
            for (int s = 0; s < SA_S; ++s) attn[tid * SA_S + s] = l[s] * rs + 1e-8f;
        }
        __syncthreads();
        if (tid < SA_S) {
            float s = 0;
            for (int n = 0; n < SA_N; ++n) s += attn[n * SA_S + tid];
            colsum[tid] = s;
        }
        __syncthreads();

        // updates (stored in lnb) = (attn/colsum)^T @ v
        if (tid < SA_S * SA_D) {
            int s = tid >> 5, d = tid & 31;
            float acc = 0;
            for (int n = 0; n < SA_N; ++n) acc += attn[n * SA_S + s] * (float)vbuf[n * 32 + d];
            lnb[tid] = acc / colsum[s];
        }
        __syncthreads();

        // GRU gates
        for (int idx = tid; idx < SA_S * 96; idx += 256) {
            int s = idx / 96, j = idx % 96;
            float a1 = bi[j], a2 = bh[j];
            for (int e = 0; e < 32; ++e) {
                a1 += lnb[s * 32 + e] * wi[j * 32 + e];
                a2 += prev[s * 32 + e] * wh[j * 32 + e];
            }
            gi[idx] = a1; gh[idx] = a2;
        }
        __syncthreads();
        if (tid < SA_S * SA_D) {
            int s = tid >> 5, d = tid & 31;
            float r  = 1.f / (1.f + expf(-(gi[s * 96 + d]      + gh[s * 96 + d])));
            float z  = 1.f / (1.f + expf(-(gi[s * 96 + 32 + d] + gh[s * 96 + 32 + d])));
            float nn = tanhf(gi[s * 96 + 64 + d] + r * gh[s * 96 + 64 + d]);
            slots[tid] = (1.f - z) * nn + z * prev[tid];
        }
        __syncthreads();

        // residual MLP: slots += relu(ln_m(slots)@m1w^T + m1b)@m2w^T + m2b
        if (tid < SA_S) {
            float s = 0;
            for (int e = 0; e < 32; ++e) s += slots[tid * 32 + e];
            float mn = s * (1.f / 32.f), vr = 0;
            for (int e = 0; e < 32; ++e) { float d = slots[tid * 32 + e] - mn; vr += d * d; }
            muS[tid] = mn; invS[tid] = rsqrtf(vr * (1.f / 32.f) + 1e-5f);
        }
        __syncthreads();
        if (tid < SA_S * SA_D) {
            int s = tid >> 5, d = tid & 31;
            lnb[tid] = (slots[tid] - muS[s]) * invS[s] * ln_m_g[d] + ln_m_b[d];
        }
        __syncthreads();
        for (int idx = tid; idx < SA_S * 64; idx += 256) {
            int s = idx / 64, h = idx % 64;
            float acc = m1b[h];
            for (int e = 0; e < 32; ++e) acc += lnb[s * 32 + e] * m1w[h * 32 + e];
            h1[idx] = acc > 0 ? acc : 0;
        }
        __syncthreads();
        if (tid < SA_S * SA_D) {
            int s = tid >> 5, d = tid & 31;
            float acc = m2b[d];
            for (int h = 0; h < 64; ++h) acc += h1[s * 64 + h] * m2w[d * 64 + h];
            slots[tid] += acc;
        }
        __syncthreads();
    }

    // mean over slots, FC 32->15
    if (tid < 15) {
        float o = fcb[tid];
        for (int d = 0; d < 32; ++d) {
            float pm = 0;
            for (int s = 0; s < SA_S; ++s) pm += slots[s * 32 + d];
            o += pm * (1.f / SA_S) * fcw[tid * 32 + d];
        }
        out[(size_t)b * 15 + tid] = o;
    }
}

// ---------------------------------------------------------------------------
// Host orchestration
// ---------------------------------------------------------------------------
enum {
    IN_X1 = 0, IN_X2 = 1, IN_NOISE = 2,
    C1A = 3, C2A = 9, C3A = 15, C1B = 21, C2B = 27, C3B = 33,
    SA_T1 = 39, SA_T2 = 45, SA_DRW = 51,
    SE_T1 = 52, SE_T2 = 58, SE_DRW = 64,
    FUS = 65,
    LN_IN_G = 71, LN_IN_B = 72, LN_S_G = 73, LN_S_B = 74, LN_M_G = 75, LN_M_B = 76,
    MUP = 77, LSIG = 78, WQ = 79, WKk = 80, WVv = 81,
    GRU_WI = 82, GRU_WH = 83, GRU_BI = 84, GRU_BH = 85,
    M1W = 86, M1B = 87, M2W = 88, M2B = 89,
    FC_W = 90, FC_B = 91
};

struct ConvPack { _Float16* apack; float* bias; int nchunks; int tilesM; };

extern "C" void kernel_launch(void* const* d_in, const int* in_sizes, int n_in,
                              void* d_out, int out_size, void* d_ws, size_t ws_size,
                              hipStream_t stream)
{
    (void)in_sizes; (void)n_in; (void)out_size; (void)ws_size;
    auto F = [&](int i) { return (const float*)d_in[i]; };

    char* ws = (char*)d_ws;
    size_t off = 0;
    auto alloc = [&](size_t bytes) -> void* {
        void* p = ws + off;
        off = (off + bytes + 255) & ~(size_t)255;
        return p;
    };

    auto prep_conv = [&](int base, int Cin, int Cout) -> ConvPack {
        int K = Cin * 9;
        int nchunks = (K + 31) / 32;
        int tilesM = (Cout + 15) / 16;
        int total = tilesM * nchunks * 512;
        _Float16* apack = (_Float16*)alloc((size_t)total * 2);
        float* bias = (float*)alloc((size_t)Cout * 4);
        pack_weights<<<(total + 255) / 256, 256, 0, stream>>>(
            F(base), F(base + 2), F(base + 5), apack, Cout, K, nchunks, total);
        fold_bias<<<(Cout + 63) / 64, 64, 0, stream>>>(
            F(base + 1), F(base + 2), F(base + 3), F(base + 4), F(base + 5), bias, Cout);
        return {apack, bias, nchunks, tilesM};
    };

    ConvPack p1a = prep_conv(C1A, 144, 128);
    ConvPack p2a = prep_conv(C2A, 128, 64);
    ConvPack p3a = prep_conv(C3A, 64, 32);
    ConvPack p1b = prep_conv(C1B, 1, 8);
    ConvPack p2b = prep_conv(C2B, 8, 16);
    ConvPack p3b = prep_conv(C3B, 16, 32);
    ConvPack pfu = prep_conv(FUS, 64, 32);

    // folded 1x1 sigmoid-block weights
    float *sa1w, *sa1b, *sa2w, *sa2b, *se1w, *se1b, *se2w, *se2b;
    auto prep1x1 = [&](int base, int Cin, int Cout, float*& wout, float*& bout) {
        wout = (float*)alloc((size_t)Cout * Cin * 4);
        bout = (float*)alloc((size_t)Cout * 4);
        fold_w1x1<<<(Cout * Cin + 255) / 256, 256, 0, stream>>>(
            F(base), F(base + 2), F(base + 5), wout, Cout, Cin);
        fold_bias<<<(Cout + 63) / 64, 64, 0, stream>>>(
            F(base + 1), F(base + 2), F(base + 3), F(base + 4), F(base + 5), bout, Cout);
    };
    prep1x1(SA_T1, 32, 16, sa1w, sa1b);
    prep1x1(SA_T2, 32, 16, sa2w, sa2b);
    prep1x1(SE_T1, 32, 32, se1w, se1b);
    prep1x1(SE_T2, 32, 32, se2w, se2b);

    // activation buffers (f16)
    _Float16* actA1 = (_Float16*)alloc((size_t)BSZ * 128 * HW * 2);
    _Float16* actA2 = (_Float16*)alloc((size_t)BSZ * 64 * HW * 2);
    _Float16* actA3 = (_Float16*)alloc((size_t)BSZ * 32 * HW * 2);
    _Float16* actB1 = (_Float16*)alloc((size_t)BSZ * 8 * HW * 2);
    _Float16* actB2 = (_Float16*)alloc((size_t)BSZ * 16 * HW * 2);
    _Float16* actB3 = (_Float16*)alloc((size_t)BSZ * 32 * HW * 2);
    _Float16* saT1  = (_Float16*)alloc((size_t)BSZ * 16 * HW * 2);
    _Float16* saT2  = (_Float16*)alloc((size_t)BSZ * 16 * HW * 2);
    _Float16* seT1  = (_Float16*)alloc((size_t)BSZ * 32 * HW * 2);
    _Float16* seT2  = (_Float16*)alloc((size_t)BSZ * 32 * HW * 2);
    _Float16* fused = (_Float16*)alloc((size_t)BSZ * 64 * HW * 2);
    _Float16* fout  = (_Float16*)alloc((size_t)BSZ * 32 * HW * 2);

    auto run_conv = [&](ConvPack& p, const float* inf, const _Float16* inh,
                        _Float16* outp, int Cin, int Cout) {
        dim3 grid(11, 1, BSZ);
        int threads = 32 * p.tilesM;
        size_t smem = (size_t)p.nchunks * 512 * sizeof(_Float16);
        conv3x3_wmma<<<grid, threads, smem, stream>>>(p.apack, p.bias, inh, inf, outp,
                                                      Cin, Cout, p.nchunks);
    };

    run_conv(p1a, (const float*)d_in[IN_X1], nullptr, actA1, 144, 128);
    run_conv(p1b, (const float*)d_in[IN_X2], nullptr, actB1, 1, 8);
    run_conv(p2a, nullptr, actA1, actA2, 128, 64);
    run_conv(p2b, nullptr, actB1, actB2, 8, 16);
    run_conv(p3a, nullptr, actA2, actA3, 64, 32);
    run_conv(p3b, nullptr, actB2, actB3, 16, 32);

    int tot16 = BSZ * 16 * HW;
    int tot32 = BSZ * 32 * HW;
    conv1x1_sig<<<(tot16 + 255) / 256, 256, 0, stream>>>(sa1w, sa1b, actA3, saT1, 32, 16, tot16);
    conv1x1_sig<<<(tot16 + 255) / 256, 256, 0, stream>>>(sa2w, sa2b, actB3, saT2, 32, 16, tot16);
    conv1x1_sig<<<(tot32 + 255) / 256, 256, 0, stream>>>(se1w, se1b, actB3, seT1, 32, 32, tot32);
    conv1x1_sig<<<(tot32 + 255) / 256, 256, 0, stream>>>(se2w, se2b, actA3, seT2, 32, 32, tot32);

    saem_gate_fuse<<<BSZ, 256, 0, stream>>>(saT1, saT2, F(SA_DRW), actA3, fused);
    seem_gate_fuse<<<BSZ, 256, 0, stream>>>(seT1, seT2, F(SE_DRW), actB3, fused);

    run_conv(pfu, nullptr, fused, fout, 64, 32);

    slot_attn_kernel<<<BSZ, 256, 0, stream>>>(
        fout, F(IN_NOISE),
        F(LN_IN_G), F(LN_IN_B), F(LN_S_G), F(LN_S_B), F(LN_M_G), F(LN_M_B),
        F(MUP), F(LSIG), F(WQ), F(WKk), F(WVv),
        F(GRU_WI), F(GRU_WH), F(GRU_BI), F(GRU_BH),
        F(M1W), F(M1B), F(M2W), F(M2B),
        F(FC_W), F(FC_B),
        (float*)d_out);
}
